// Conv2_5d_2104533975528
// MI455X (gfx1250) — matrix operand order repaired and checked by reading, not compiled, not run on hardware
//
#include <hip/hip_runtime.h>

typedef __attribute__((ext_vector_type(16))) _Float16 v16h;
typedef __attribute__((ext_vector_type(8)))  float    v8f;
typedef __attribute__((ext_vector_type(8)))  unsigned v8u;
typedef __attribute__((ext_vector_type(4)))  unsigned v4u;

#define B_   4
#define CIN  64
#define COUT 64
#define H_   128
#define W_   128
#define KS   3

#define XT   66      // 64 pixels + 2 halo columns
#define CPAD 72      // padded channel stride in halfs (bank-conflict avoidance)

// ---------------------------------------------------------------------------
// Prep: swizzle f32 weights (3, Cout, Cin, 3, 3) into f16 B-fragments laid out
// in this code's contiguous-K per-lane order (same as the A loads):
//   fragment fr = q*4 + ntile, q = (h*9 + ky*3 + kx)*2 + ccat  (K-chunk of 32 c)
//   lane < 16 : column n=lane,    halfs j=0..15 -> K=j        (c = ccat*32 + j)
//   lane >=16 : column n=lane-16, halfs j=0..15 -> K=16+j
// ---------------------------------------------------------------------------
__global__ __launch_bounds__(256) void swizzle_weights(const float* __restrict__ w,
                                                       _Float16* __restrict__ wswz) {
    int idx = blockIdx.x * 256 + threadIdx.x;      // [0, 216*512)
    if (idx >= 216 * 512) return;
    int j    = idx & 15;
    int lane = (idx >> 4) & 31;
    int fr   = idx >> 9;
    int nt   = fr & 3;
    int q    = fr >> 2;          // [0,54)
    int ccat = q & 1;
    int t    = q >> 1;           // h*9 + ky*3 + kx
    int kx   = t % 3;
    int ky   = (t / 3) % 3;
    int h    = t / 9;
    int o    = nt * 16 + (lane & 15);
    int c    = ccat * 32 + (lane >> 4) * 16 + j;
    float v  = w[(((h * COUT + o) * CIN + c) * KS + ky) * KS + kx];
    wswz[idx] = (_Float16)v;
}

// ---------------------------------------------------------------------------
// Main: implicit-GEMM 2.5D conv. Block = 128 thr = 4 waves; tile = 64 pixels
// of one row (b,y) x 64 Cout. Wave wv owns Cout[wv*16..+15] and 4 M-subtiles.
// K = 3 depth planes * 9 taps * 64 Cin = 1728; masks gate whole 32-wide
// K-chunks of A with a branch-free vector AND.
// ---------------------------------------------------------------------------
__global__ __launch_bounds__(128) void conv25d_wmma(
    const float* __restrict__ inp, const float* __restrict__ dep,
    const _Float16* __restrict__ wswz, const float* __restrict__ bias,
    const float* __restrict__ fptr, float* __restrict__ out)
{
    __shared__ __attribute__((aligned(16))) _Float16 sIn[3 * XT * CPAD]; // 28.5 KB
    __shared__ float sD[3 * XT];

    const int tid  = threadIdx.x;
    const int xseg = blockIdx.x & 1;
    const int y    = (blockIdx.x >> 1) & (H_ - 1);
    const int b    = blockIdx.x >> 8;
    const int x0   = xseg * 64;

    // ---- stage depth halo (zero padded) ----
    for (int e = tid; e < 3 * XT; e += 128) {
        int row = e / XT, xi = e % XT;
        int yy = y + row - 1, xx = x0 + xi - 1;
        float v = 0.f;
        if (yy >= 0 && yy < H_ && xx >= 0 && xx < W_)
            v = dep[(b * H_ + yy) * W_ + xx];
        sD[e] = v;
    }
    // ---- stage input halo tile as f16, layout [row][xi][c] ----
    for (int e = tid; e < CIN * 3 * XT; e += 128) {
        int c   = e / (3 * XT);
        int rem = e % (3 * XT);
        int row = rem / XT, xi = rem % XT;
        int yy = y + row - 1, xx = x0 + xi - 1;
        float v = 0.f;
        if (yy >= 0 && yy < H_ && xx >= 0 && xx < W_)
            v = inp[((b * CIN + c) * H_ + yy) * W_ + xx];
        sIn[(row * XT + xi) * CPAD + c] = (_Float16)v;
    }
    __syncthreads();

    const int lane = tid & 31;
    const int wv   = tid >> 5;       // wave id == Cout tile
    const int p    = lane & 15;      // pixel row within M-subtile (A) / column n (B,D)
    const int g    = lane >> 4;
    const float f  = fptr[0];

    // ---- per-lane depth-bin masks: 27 bits (h,ky,kx) per M-subtile ----
    unsigned mbits[4];
#pragma unroll
    for (int mt = 0; mt < 4; ++mt) {
        float d0  = sD[XT + mt * 16 + p + 1];     // center depth (row 1, kx=1)
        float s0  = d0 / f;
        float hlf = 0.5f * s0;
        unsigned bits = 0;
#pragma unroll
        for (int h = 0; h < 3; ++h) {
            float z0 = d0 + ((float)h - 1.0f) * s0;
            float lo = z0 - hlf, hi = z0 + hlf;
#pragma unroll
            for (int ky = 0; ky < 3; ++ky)
#pragma unroll
                for (int kx = 0; kx < 3; ++kx) {
                    float dw = sD[ky * XT + mt * 16 + p + kx];
                    if (dw >= lo && dw < hi) bits |= 1u << (h * 9 + ky * 3 + kx);
                }
        }
        mbits[mt] = bits;
    }

    v8f acc[4] = {v8f{}, v8f{}, v8f{}, v8f{}};
    const v16h* wv16 = (const v16h*)wswz;

#pragma unroll
    for (int ky = 0; ky < 3; ++ky)
#pragma unroll
    for (int kx = 0; kx < 3; ++kx)
#pragma unroll
    for (int ccat = 0; ccat < 2; ++ccat) {
        // A fragments (unmasked), reused across the 3 depth planes.
        // 16-bit A 16x32 layout: lane g holds K = g*16+0..15 (matches B swizzle).
        v8u araw[4];
#pragma unroll
        for (int mt = 0; mt < 4; ++mt) {
            int base = (ky * XT + (mt * 16 + p + kx)) * CPAD + ccat * 32 + g * 16;
            v4u lo = *(const v4u*)&sIn[base];
            v4u hi = *(const v4u*)&sIn[base + 8];
            araw[mt] = __builtin_shufflevector(lo, hi, 0, 1, 2, 3, 4, 5, 6, 7);
        }
#pragma unroll
        for (int h = 0; h < 3; ++h) {
            const int q = (h * 9 + ky * 3 + kx) * 2 + ccat;
            const v16h* bp = wv16 + (q * 4 + wv) * 32 + lane;
            v16h bw = *bp;
            __builtin_prefetch((const void*)(bp + 4 * 32), 0, 1); // next q's fragment
            const int bitidx = h * 9 + ky * 3 + kx;
#pragma unroll
            for (int mt = 0; mt < 4; ++mt) {
                // branch-free gate: all-ones / all-zeros word, vector AND
                unsigned m32 = 0u - ((mbits[mt] >> bitidx) & 1u);
                v8u am = araw[mt] & m32;
                acc[mt] = __builtin_amdgcn_wmma_f32_16x16x32_f16(
                    /*neg_a=*/false, __builtin_bit_cast(v16h, am),
                    /*neg_b=*/false, bw,
                    /*c_mod=*/(short)0, acc[mt],
                    /*reuse_a=*/false, /*reuse_b=*/false);
            }
        }
    }

    // ---- epilogue: D layout -> lane holds column o, rows M = r + 8g (contiguous x)
    int o = wv * 16 + p;
    float bo = bias[o];
#pragma unroll
    for (int mt = 0; mt < 4; ++mt) {
        int xw = x0 + mt * 16 + g * 8;
        float* dst = &out[((b * COUT + o) * H_ + y) * W_ + xw];
        float4 v0 = {acc[mt][0] + bo, acc[mt][1] + bo, acc[mt][2] + bo, acc[mt][3] + bo};
        float4 v1 = {acc[mt][4] + bo, acc[mt][5] + bo, acc[mt][6] + bo, acc[mt][7] + bo};
        *(float4*)dst = v0;
        *(float4*)(dst + 4) = v1;
    }
}

extern "C" void kernel_launch(void* const* d_in, const int* in_sizes, int n_in,
                              void* d_out, int out_size, void* d_ws, size_t ws_size,
                              hipStream_t stream) {
    (void)in_sizes; (void)n_in; (void)out_size; (void)ws_size;
    const float* inputs = (const float*)d_in[0];
    const float* depth  = (const float*)d_in[1];
    const float* weight = (const float*)d_in[2];
    const float* bias   = (const float*)d_in[3];
    const float* fsc    = (const float*)d_in[4];
    float* out          = (float*)d_out;
    _Float16* wswz      = (_Float16*)d_ws;   // needs 216*512*2 = 221184 bytes

    swizzle_weights<<<dim3(432), dim3(256), 0, stream>>>(weight, wswz);
    conv25d_wmma<<<dim3(B_ * H_ * 2), dim3(128), 0, stream>>>(
        inputs, depth, wswz, bias, fsc, out);
}
